// TDTFLayer_43147241456138
// MI455X (gfx1250) — matrix operand order repaired and checked by reading, not compiled, not run on hardware
//
#include <hip/hip_runtime.h>
#include <math.h>

// ---------------------------------------------------------------- constants
#define B_SZ   2
#define T_TOK  2048
#define DMODEL 2048
#define HEADS  16
#define HDIM   128
#define FFN    8192
#define NTOK   (B_SZ * T_TOK)

typedef unsigned short u16;
typedef unsigned int   u32;
typedef unsigned long long u64;
typedef __attribute__((ext_vector_type(16))) __bf16          v16bf;
typedef __attribute__((ext_vector_type(16))) unsigned short  v16u;
typedef __attribute__((ext_vector_type(8)))  float           v8f;

// ---------------------------------------------------------------- bf16 utils
__device__ __forceinline__ u16 f2bf(float f) {
  u32 u = __builtin_bit_cast(u32, f);
  u32 r = u + 0x7FFFu + ((u >> 16) & 1u);   // round-to-nearest-even
  return (u16)(r >> 16);
}
__device__ __forceinline__ float bf2f(u16 h) {
  u32 u = ((u32)h) << 16;
  return __builtin_bit_cast(float, u);
}

// ---------------------------------------------------------------- async copy (CDNA5 ASYNCcnt path)
// 16B global -> LDS, tracked by ASYNCcnt (CDNA5 ISA).
// LDS address = low 32 bits of the flat shared pointer (aperture truncation).
__device__ __forceinline__ void async_copy_b128(void* lds_ptr, const void* gptr) {
  u32 lds = (u32)(size_t)lds_ptr;
  u64 ga  = (u64)(size_t)gptr;
  asm volatile("global_load_async_to_lds_b128 %0, %1, off"
               :: "v"(lds), "v"(ga) : "memory");
}
__device__ __forceinline__ void async_wait0() {
  asm volatile("s_wait_asynccnt 0x0" ::: "memory");
}

// ---------------------------------------------------------------- WMMA core
__device__ __forceinline__ v8f wmma_bf16(v16u a, v16u b, v8f c) {
  return __builtin_amdgcn_wmma_f32_16x16x32_bf16(
      false, __builtin_bit_cast(v16bf, a),
      false, __builtin_bit_cast(v16bf, b),
      (short)0, c, false, false);
}

// A fragment: 16x32 bf16, row-major source (lda in elements).
// lanes 0-15: M=lane, K 0..15 ; lanes 16-31: M=lane-16, K 16..31
// -> two contiguous 16B runs per lane (2x ds_load_b128)
__device__ __forceinline__ v16u frag_a(const u16* base, int lda, int lane) {
  const int half = lane >> 4;
  const int m    = lane & 15;
  const u16* row = base + m * lda + 16 * half;
  v16u r;
#pragma unroll
  for (int e = 0; e < 8; ++e)  r[e]     = row[e];        // K 0..7 (+16*half)
#pragma unroll
  for (int e = 0; e < 8; ++e)  r[8 + e] = row[8 + e];    // K 8..15 (+16*half)
  return r;
}

// 16 consecutive u16 -> one fragment register set (2x ds_load_b128)
__device__ __forceinline__ v16u frag_contig(const u16* p) {
  v16u r;
#pragma unroll
  for (int e = 0; e < 16; ++e) r[e] = p[e];
  return r;
}

// B fragment from transposed storage (N-major rows, element B[k][n] = buf[n*ld + k0+e])
__device__ __forceinline__ v16u frag_bt(const u16* base, int ld, int lane) {
  const int n  = lane & 15;
  const int k0 = (lane >> 4) * 16;
  return frag_contig(base + n * ld + k0);
}

// ---------------------------------------------------------------- GEMM
// C[M,N] = act(A[M,K](bf16) * B_swizzled + bias) + residual ; optional bf16 copy
// Block tile 128x128x32, double-buffered LDS, fully-async staging pipeline.
// B weights are pre-swizzled in global memory into fragment-major 32x128 tiles:
//   tile(kt,nt) base = (kt*(N/128)+nt)*4096 ; element (c,kk,nl) at (c*128+nl)*16+kk
//   where c=(k>>4)&1, kk=k&15, nl=n&127.
#define BM 128
#define BN 128
#define BK 32
#define LDA_S 40              // 32 + 8 pad: stride 20 words -> conflict-free rows
#define ATILE (BM * LDA_S)    // 5120 u16
#define BTILE (2 * BN * 16)   // 4096 u16

__device__ __forceinline__ void gemm_stage(const u16* __restrict__ A,
                                           const u16* __restrict__ Bsw,
                                           u16* sAp, u16* sBp,
                                           int row0, int col0, int ntcols,
                                           int K, int kt, int tid) {
  const int k0 = kt << 5;
#pragma unroll
  for (int r = 0; r < 2; ++r) {          // A tile 128x32, async
    int idx = (tid + r * 256) * 8;
    int ar = idx >> 5, ac = idx & 31;
    async_copy_b128(&sAp[ar * LDA_S + ac], &A[(size_t)(row0 + ar) * K + k0 + ac]);
  }
  const u16* Btile = Bsw + ((size_t)kt * ntcols + (col0 >> 7)) * BTILE;
#pragma unroll
  for (int r = 0; r < 2; ++r) {          // B tile (pre-swizzled, contiguous), async
    int idx = (tid + r * 256) * 8;
    async_copy_b128(&sBp[idx], &Btile[idx]);
  }
}

__global__ __launch_bounds__(256)
void gemm_bf16_kernel(const u16* __restrict__ A, const u16* __restrict__ Bsw,
                      float* __restrict__ C, u16* __restrict__ Cbf,
                      const float* __restrict__ bias,
                      const float* __restrict__ residual,
                      int M, int N, int K, int act) {
  __shared__ u16 sA[2][ATILE];
  __shared__ u16 sBt[2][BTILE];

  const int tid  = threadIdx.x;
  const int lane = tid & 31;
  const int wave = tid >> 5;
  const int wr   = wave >> 1;                // 0..3 : 32-row strip
  const int wc   = wave & 1;                 // 0..1 : 64-col strip
  const int row0 = blockIdx.y * BM;
  const int col0 = blockIdx.x * BN;
  const int ntiles = K >> 5;
  const int ntcols = N >> 7;

  v8f acc[2][4];
#pragma unroll
  for (int i = 0; i < 2; ++i)
#pragma unroll
    for (int j = 0; j < 4; ++j)
#pragma unroll
      for (int e = 0; e < 8; ++e) acc[i][j][e] = 0.0f;

  gemm_stage(A, Bsw, sA[0], sBt[0], row0, col0, ntcols, K, 0, tid);
  async_wait0();
  __syncthreads();

  for (int kt = 0; kt < ntiles; ++kt) {
    const int p = kt & 1;
    if (kt + 1 < ntiles)     // overlap next tile's HBM fetch with WMMA issue
      gemm_stage(A, Bsw, sA[p ^ 1], sBt[p ^ 1], row0, col0, ntcols, K, kt + 1, tid);
    if (kt + 2 < ntiles)     // prefetch B tile k+2 into L2 (global_prefetch_b8)
      __builtin_prefetch(Bsw + ((size_t)(kt + 2) * ntcols + (col0 >> 7)) * BTILE + tid * 16, 0, 1);

    v16u af[2], bfr[4];
#pragma unroll
    for (int i = 0; i < 2; ++i)
      af[i] = frag_a(&sA[p][(wr * 32 + i * 16) * LDA_S], LDA_S, lane);
#pragma unroll
    for (int j = 0; j < 4; ++j)
      bfr[j] = frag_contig(&sBt[p][(((lane >> 4) * BN) + wc * 64 + j * 16 + (lane & 15)) * 16]);
#pragma unroll
    for (int i = 0; i < 2; ++i)
#pragma unroll
      for (int j = 0; j < 4; ++j)
        acc[i][j] = wmma_bf16(af[i], bfr[j], acc[i][j]);

    async_wait0();           // next buffer filled; this buffer free to overwrite
    __syncthreads();
  }

  const int half = lane >> 4;
  const int nl   = lane & 15;
#pragma unroll
  for (int i = 0; i < 2; ++i)
#pragma unroll
    for (int j = 0; j < 4; ++j)
#pragma unroll
      for (int e = 0; e < 8; ++e) {
        int gr = row0 + wr * 32 + i * 16 + e + 8 * half;
        int gc = col0 + wc * 64 + j * 16 + nl;
        float v = acc[i][j][e];
        if (bias)     v += bias[gc];
        if (act == 1) v = v / (1.0f + __expf(-v));     // SiLU
        if (residual) v += residual[(size_t)gr * N + gc];
        C[(size_t)gr * N + gc] = v;
        if (Cbf) Cbf[(size_t)gr * N + gc] = f2bf(v);
      }
}

// ---------------------------------------------------------------- RMSNorm
__global__ __launch_bounds__(256)
void rmsnorm_kernel(const float* __restrict__ x, const float* __restrict__ w,
                    u16* __restrict__ out) {
  const int row = blockIdx.x;
  const float* xr = x + (size_t)row * DMODEL;
  __shared__ float red[256];
  float s = 0.f;
  for (int i = threadIdx.x; i < DMODEL; i += 256) { float v = xr[i]; s += v * v; }
  red[threadIdx.x] = s; __syncthreads();
  for (int off = 128; off > 0; off >>= 1) {
    if (threadIdx.x < off) red[threadIdx.x] += red[threadIdx.x + off];
    __syncthreads();
  }
  const float inv = rsqrtf(red[0] / (float)DMODEL + 1e-6f);
  u16* orow = out + (size_t)row * DMODEL;
  for (int i = threadIdx.x; i < DMODEL; i += 256) orow[i] = f2bf(xr[i] * inv * w[i]);
}

// ---------------------------------------------------------------- converts
__global__ void f32_to_bf16_kernel(const float* __restrict__ in,
                                   u16* __restrict__ out, size_t n) {
  size_t i = (size_t)blockIdx.x * blockDim.x + threadIdx.x;
  if (i < n) out[i] = f2bf(in[i]);
}

// weight convert f32 -> bf16 with fragment-major 32x128 tile swizzle (see GEMM)
__global__ void cvt_swizzleB_kernel(const float* __restrict__ in,
                                    u16* __restrict__ out, int K, int N) {
  size_t i = (size_t)blockIdx.x * blockDim.x + threadIdx.x;
  if (i >= (size_t)K * N) return;
  int k = (int)(i / N), n = (int)(i % N);
  int kt = k >> 5, c = (k >> 4) & 1, kk = k & 15;
  int nt = n >> 7, nl = n & 127;
  size_t dst = ((size_t)kt * (N >> 7) + nt) * BTILE + ((size_t)(c * 128 + nl) * 16) + kk;
  out[dst] = f2bf(in[i]);
}

// ---------------------------------------------------------------- RoPE / relayout
// mode 1: RoPE, out layout [B,H,T,HD]  (Q,K)
// mode 0: plain transpose, out layout [B,H,HD,T]  (V, row-contiguous key tiles)
__global__ __launch_bounds__(128)
void rope_kernel(const float* __restrict__ in, u16* __restrict__ out, int mode) {
  const int t = blockIdx.x, h = blockIdx.y, b = blockIdx.z;
  const int d = threadIdx.x;                       // 0..127
  const float* src = in + ((size_t)b * T_TOK + t) * (HEADS * HDIM) + h * HDIM;
  float v = src[d];
  if (mode == 1) {
    int i = d & 63;
    float freq = __powf(10000.0f, -2.0f * (float)i / (float)HDIM);
    float ang = (float)t * freq;
    float c = __cosf(ang), s = __sinf(ang);
    float other = (d < 64) ? -src[d + 64] : src[d - 64];
    float o = v * c + other * s;
    out[(((size_t)b * HEADS + h) * T_TOK + t) * HDIM + d] = f2bf(o);
  } else {
    out[(((size_t)b * HEADS + h) * HDIM + d) * T_TOK + t] = f2bf(v);
  }
}

// ---------------------------------------------------------------- flash attention
// 64-query tile/block; 8 waves = 4 row-groups x 2 HD-halves; key tile = 32.
// Q,K in [B,H,T,HD]; V in [B,H,HD,T] so all tiles stage via async b128 copies.
#define QT 64
#define KT 32
#define LDQ_S 136   // 128 + 8 pad
#define LDV_S 40    // 32 + 8 pad (transposed V rows)
#define LDP 40

__global__ __launch_bounds__(256)
void flash_attn_kernel(const u16* __restrict__ Q, const u16* __restrict__ Kb,
                       const u16* __restrict__ Vt, u16* __restrict__ ctx) {
  __shared__ u16   sQ[QT * LDQ_S];
  __shared__ u16   sK[KT * LDQ_S];
  __shared__ u16   sVt[HDIM * LDV_S];   // transposed: sVt[hd][key]
  __shared__ float sS[QT * KT];
  __shared__ u16   sP[QT * LDP];
  __shared__ float sM[QT], sL[QT], sScale[QT];

  const int tid  = threadIdx.x;
  const int lane = tid & 31;
  const int wave = tid >> 5;
  const int rg   = wave & 3;        // row group (16 query rows)
  const int hh   = wave >> 2;       // HD half (64 output cols)
  const int q0   = blockIdx.x * QT;
  const int h    = blockIdx.y;
  const int b    = blockIdx.z;

  const u16* Qh  = Q  + (((size_t)b * HEADS + h) * T_TOK) * HDIM;
  const u16* Kh  = Kb + (((size_t)b * HEADS + h) * T_TOK) * HDIM;
  const u16* Vth = Vt + (((size_t)b * HEADS + h) * HDIM) * T_TOK;

#pragma unroll
  for (int r = 0; r < 4; ++r) {           // Q tile 64x128: async copies
    int idx = (tid + r * 256) * 8;
    int qr = idx >> 7, qc = idx & 127;
    async_copy_b128(&sQ[qr * LDQ_S + qc], &Qh[(size_t)(q0 + qr) * HDIM + qc]);
  }
  if (tid < QT) { sM[tid] = -3.0e38f; sL[tid] = 0.f; }

  v8f acc[4];
#pragma unroll
  for (int j = 0; j < 4; ++j)
#pragma unroll
    for (int e = 0; e < 8; ++e) acc[j][e] = 0.f;

  const int   ntiles = (q0 + QT - 1) / KT + 1;     // causal bound
  const float scl    = 0.08838834764831845f;       // 1/sqrt(128)
  async_wait0();
  __syncthreads();

  for (int kt = 0; kt < ntiles; ++kt) {
    const int k0 = kt * KT;
#pragma unroll
    for (int r = 0; r < 2; ++r) {        // K tile 32x128: async copies
      int idx = (tid + r * 256) * 8;
      int kr = idx >> 7, kc = idx & 127;
      async_copy_b128(&sK[kr * LDQ_S + kc], &Kh[(size_t)(k0 + kr) * HDIM + kc]);
    }
#pragma unroll
    for (int r = 0; r < 2; ++r) {        // V^T tile 128x32: async copies
      int idx = (tid + r * 256) * 8;
      int vr = idx >> 5, vk = idx & 31;
      async_copy_b128(&sVt[vr * LDV_S + vk], &Vth[(size_t)vr * T_TOK + k0 + vk]);
    }
    async_wait0();
    __syncthreads();

    if (hh == 0) {                        // S = Q K^T (waves 0..3), causal mask
#pragma unroll
      for (int jn = 0; jn < 2; ++jn) {
        v8f s;
#pragma unroll
        for (int e = 0; e < 8; ++e) s[e] = 0.f;
#pragma unroll
        for (int kc = 0; kc < 4; ++kc) {
          v16u a  = frag_a (&sQ[(rg * 16) * LDQ_S + kc * 32], LDQ_S, lane);
          v16u bq = frag_bt(&sK[(jn * 16) * LDQ_S + kc * 32], LDQ_S, lane);
          s = wmma_bf16(a, bq, s);
        }
        const int half = lane >> 4, nl = lane & 15;
        const int gk = k0 + jn * 16 + nl;
#pragma unroll
        for (int e = 0; e < 8; ++e) {
          int lrow = e + 8 * half;
          int gq = q0 + rg * 16 + lrow;
          float v = s[e] * scl;
          if (gk > gq) v = -1.0e9f;
          sS[(rg * 16 + lrow) * KT + jn * 16 + nl] = v;
        }
      }
    }
    __syncthreads();

    if (tid < QT) {                       // online softmax, one row per thread
      float m_old = sM[tid];
      float mmax = m_old;
      for (int c = 0; c < KT; ++c) mmax = fmaxf(mmax, sS[tid * KT + c]);
      float resc = __expf(m_old - mmax);
      float lsum = sL[tid] * resc;
      for (int c = 0; c < KT; ++c) {
        float p = __expf(sS[tid * KT + c] - mmax);
        lsum += p;
        sP[tid * LDP + c] = f2bf(p);
      }
      sM[tid] = mmax; sL[tid] = lsum; sScale[tid] = resc;
    }
    __syncthreads();

    {                                     // rescale + O += P V
      const int half = lane >> 4;
#pragma unroll
      for (int j = 0; j < 4; ++j)
#pragma unroll
        for (int e = 0; e < 8; ++e)
          acc[j][e] *= sScale[rg * 16 + e + 8 * half];
      v16u a = frag_a(&sP[(rg * 16) * LDP], LDP, lane);
#pragma unroll
      for (int j = 0; j < 4; ++j) {
        v16u bv = frag_bt(&sVt[(hh * 64 + j * 16) * LDV_S], LDV_S, lane);
        acc[j] = wmma_bf16(a, bv, acc[j]);
      }
    }
    __syncthreads();
  }

  {                                       // normalize + write [B,T,H*HD] bf16
    const int half = lane >> 4, nl = lane & 15;
#pragma unroll
    for (int j = 0; j < 4; ++j)
#pragma unroll
      for (int e = 0; e < 8; ++e) {
        int lrow = rg * 16 + e + 8 * half;
        int t = q0 + lrow;
        int col = hh * 64 + j * 16 + nl;
        float invl = 1.0f / sL[lrow];
        ctx[((size_t)b * T_TOK + t) * DMODEL + h * HDIM + col] = f2bf(acc[j][e] * invl);
      }
  }
}

// ---------------------------------------------------------------- elementwise / reductions
__global__ void swiglu_mul_kernel(const float* __restrict__ g,
                                  const float* __restrict__ u,
                                  u16* __restrict__ out, size_t n) {
  size_t i = (size_t)blockIdx.x * blockDim.x + threadIdx.x;
  if (i < n) out[i] = f2bf(g[i] * u[i]);    // g already has SiLU applied
}

__global__ __launch_bounds__(256)
void shift_kernel(const float* __restrict__ xpost, u16* __restrict__ prev) {
  const int tok = blockIdx.x;
  const int t = tok % T_TOK;
  u16* dst = prev + (size_t)tok * DMODEL;
  if (t == 0) {
    for (int i = threadIdx.x; i < DMODEL; i += 256) dst[i] = 0;
  } else {
    const float* src = xpost + (size_t)(tok - 1) * DMODEL;
    for (int i = threadIdx.x; i < DMODEL; i += 256) dst[i] = f2bf(src[i]);
  }
}

__global__ __launch_bounds__(256)
void router_gemv_kernel(const float* __restrict__ x, const float* __restrict__ w,
                        const float* __restrict__ b, float* __restrict__ scores) {
  const int tok = blockIdx.x;
  const float* xr = x + (size_t)tok * DMODEL;
  __shared__ float red[256];
  float s = 0.f;
  for (int i = threadIdx.x; i < DMODEL; i += 256) s += xr[i] * w[i];
  red[threadIdx.x] = s; __syncthreads();
  for (int off = 128; off > 0; off >>= 1) {
    if (threadIdx.x < off) red[threadIdx.x] += red[threadIdx.x + off];
    __syncthreads();
  }
  if (threadIdx.x == 0) scores[tok] = red[0] + b[0];
}

__global__ __launch_bounds__(256)
void tpn_err_kernel(const float* __restrict__ xpost, const float* __restrict__ xorig,
                    const float* __restrict__ pred, float* __restrict__ err,
                    float* __restrict__ tpn_acc) {
  const int tok = blockIdx.x;
  const size_t base = (size_t)tok * DMODEL;
  __shared__ float red[256];
  float s = 0.f;
  for (int i = threadIdx.x; i < DMODEL; i += 256) {
    float a = xpost[base + i] - xorig[base + i];
    float d = pred[base + i] - a;
    s += d * d;
  }
  red[threadIdx.x] = s; __syncthreads();
  for (int off = 128; off > 0; off >>= 1) {
    if (threadIdx.x < off) red[threadIdx.x] += red[threadIdx.x + off];
    __syncthreads();
  }
  if (threadIdx.x == 0) {
    err[tok] = red[0] / (float)DMODEL;
    atomicAdd(tpn_acc, red[0]);
  }
}

__global__ __launch_bounds__(256)
void err_stats_kernel(const float* __restrict__ err, float* __restrict__ stats) {
  __shared__ float rs[256], rq[256];
  float s = 0.f, q = 0.f;
  for (int i = threadIdx.x; i < NTOK; i += 256) { float e = err[i]; s += e; q += e * e; }
  rs[threadIdx.x] = s; rq[threadIdx.x] = q; __syncthreads();
  for (int off = 128; off > 0; off >>= 1) {
    if (threadIdx.x < off) { rs[threadIdx.x] += rs[threadIdx.x + off];
                             rq[threadIdx.x] += rq[threadIdx.x + off]; }
    __syncthreads();
  }
  if (threadIdx.x == 0) {
    float mean = rs[0] / (float)NTOK;
    float var = rq[0] / (float)NTOK - mean * mean;
    stats[0] = mean;
    stats[1] = sqrtf(fmaxf(var, 0.f));
  }
}

__global__ void router_final_kernel(const float* __restrict__ err,
                                    const float* __restrict__ scores,
                                    const float* __restrict__ stats,
                                    float* __restrict__ g_out, float* __restrict__ tgt_out,
                                    float* __restrict__ prob_out,
                                    float* __restrict__ closs_acc) {
  const int i = blockIdx.x * blockDim.x + threadIdx.x;
  if (i >= NTOK) return;
  const float mean = stats[0], stdv = stats[1];
  float g = 1.f / (1.f + __expf(-((err[i] - mean) / (stdv + 1e-6f))));
  float tgt = g > 0.5f ? 1.f : 0.f;
  float s = scores[i];
  float prob = 1.f / (1.f + __expf(-s));
  float l = fmaxf(s, 0.f) - s * tgt + log1pf(__expf(-fabsf(s)));
  g_out[i] = g; tgt_out[i] = tgt; prob_out[i] = prob;
  atomicAdd(closs_acc, l);
}

__global__ void finalize_kernel(const float* __restrict__ tpn_acc,
                                const float* __restrict__ closs_acc,
                                float* __restrict__ tpn_out,
                                float* __restrict__ closs_out) {
  if (threadIdx.x == 0 && blockIdx.x == 0) {
    *tpn_out = *tpn_acc / (float)((size_t)NTOK * DMODEL);
    *closs_out = *closs_acc / (float)NTOK;
  }
}

// ================================================================ launcher
extern "C" void kernel_launch(void* const* d_in, const int* in_sizes, int n_in,
                              void* d_out, int out_size, void* d_ws, size_t ws_size,
                              hipStream_t stream) {
  (void)in_sizes; (void)n_in; (void)out_size; (void)ws_size;

  const float* hidden = (const float*)d_in[0];
  const float* q_w  = (const float*)d_in[1];  const float* q_b  = (const float*)d_in[2];
  const float* k_w  = (const float*)d_in[3];  const float* k_b  = (const float*)d_in[4];
  const float* v_w  = (const float*)d_in[5];  const float* v_b  = (const float*)d_in[6];
  const float* o_w  = (const float*)d_in[7];
  const float* ln1  = (const float*)d_in[8];  const float* ln2  = (const float*)d_in[9];
  const float* g_w  = (const float*)d_in[10]; const float* u_w  = (const float*)d_in[11];
  const float* dn_w = (const float*)d_in[12];
  const float* t1_w = (const float*)d_in[13]; const float* t1_b = (const float*)d_in[14];
  const float* t2_w = (const float*)d_in[15]; const float* t2_b = (const float*)d_in[16];
  const float* r_w  = (const float*)d_in[17]; const float* r_b  = (const float*)d_in[18];

  const size_t DD = (size_t)DMODEL * DMODEL;
  const size_t DF = (size_t)DMODEL * FFN;
  const size_t MD = (size_t)NTOK * DMODEL;
  const size_t MF = (size_t)NTOK * FFN;

  char* ws = (char*)d_ws;
  size_t off = 0;
  auto alloc = [&](size_t bytes) -> void* {
    void* p = ws + off; off += (bytes + 255) & ~(size_t)255; return p;
  };

  u16* wq  = (u16*)alloc(DD * 2);  u16* wk  = (u16*)alloc(DD * 2);
  u16* wv  = (u16*)alloc(DD * 2);  u16* wo  = (u16*)alloc(DD * 2);
  u16* wg  = (u16*)alloc(DF * 2);  u16* wu  = (u16*)alloc(DF * 2);
  u16* wd  = (u16*)alloc(DF * 2);
  u16* wt1 = (u16*)alloc(DD * 2);  u16* wt2 = (u16*)alloc(DD * 2);

  u16*   hnorm = (u16*)alloc(MD * 2);
  float* qf    = (float*)alloc(MD * 4);
  float* kf    = (float*)alloc(MD * 4);
  float* vf    = (float*)alloc(MD * 4);
  u16*   qr    = (u16*)alloc(MD * 2);
  u16*   kr    = (u16*)alloc(MD * 2);
  u16*   vtr   = (u16*)alloc(MD * 2);
  u16*   ctxb  = (u16*)alloc(MD * 2);
  float* xf    = (float*)alloc(MD * 4);
  u16*   h2b   = (u16*)alloc(MD * 2);
  float* gatef = (float*)alloc(MF * 4);
  float* upf   = (float*)alloc(MF * 4);
  u16*   mulb  = (u16*)alloc(MF * 2);
  u16*   prevb = (u16*)alloc(MD * 2);
  float* t1f   = (float*)alloc(MD * 4);
  u16*   t1b16 = (u16*)alloc(MD * 2);
  float* predf = (float*)alloc(MD * 4);
  float* scoresf = (float*)alloc((size_t)NTOK * 4);
  float* errf    = (float*)alloc((size_t)NTOK * 4);
  float* accums  = (float*)alloc(2 * 4);   // [tpn_acc, closs_acc]
  float* statsb  = (float*)alloc(2 * 4);   // [mean, std]

  float* out    = (float*)d_out;
  float* xpost  = out;                     // [B,T,D]
  float* tpn_o  = out + MD;
  float* clo_o  = tpn_o + 1;
  float* g_o    = clo_o + 1;
  float* tgt_o  = g_o + NTOK;
  float* prob_o = tgt_o + NTOK;

  hipMemsetAsync(accums, 0, 2 * sizeof(float), stream);

  auto cvtB = [&](const float* src, u16* dst, int K, int N) {
    size_t n = (size_t)K * N;
    cvt_swizzleB_kernel<<<(unsigned)((n + 255) / 256), 256, 0, stream>>>(src, dst, K, N);
  };
  cvtB(q_w, wq, DMODEL, DMODEL); cvtB(k_w, wk, DMODEL, DMODEL);
  cvtB(v_w, wv, DMODEL, DMODEL); cvtB(o_w, wo, DMODEL, DMODEL);
  cvtB(g_w, wg, DMODEL, FFN);    cvtB(u_w, wu, DMODEL, FFN);
  cvtB(dn_w, wd, FFN, DMODEL);
  cvtB(t1_w, wt1, DMODEL, DMODEL); cvtB(t2_w, wt2, DMODEL, DMODEL);

  // ---- attention block ----
  rmsnorm_kernel<<<NTOK, 256, 0, stream>>>(hidden, ln1, hnorm);

  dim3 gD(DMODEL / BN, NTOK / BM);   // N=2048 GEMMs
  dim3 gF(FFN / BN, NTOK / BM);      // N=8192 GEMMs
  gemm_bf16_kernel<<<gD, 256, 0, stream>>>(hnorm, wq, qf, nullptr, q_b, nullptr,
                                           NTOK, DMODEL, DMODEL, 0);
  gemm_bf16_kernel<<<gD, 256, 0, stream>>>(hnorm, wk, kf, nullptr, k_b, nullptr,
                                           NTOK, DMODEL, DMODEL, 0);
  gemm_bf16_kernel<<<gD, 256, 0, stream>>>(hnorm, wv, vf, nullptr, v_b, nullptr,
                                           NTOK, DMODEL, DMODEL, 0);

  dim3 gR(T_TOK, HEADS, B_SZ);
  rope_kernel<<<gR, 128, 0, stream>>>(qf, qr, 1);
  rope_kernel<<<gR, 128, 0, stream>>>(kf, kr, 1);
  rope_kernel<<<gR, 128, 0, stream>>>(vf, vtr, 0);   // V -> [B,H,HD,T]

  dim3 gA(T_TOK / QT, HEADS, B_SZ);
  flash_attn_kernel<<<gA, 256, 0, stream>>>(qr, kr, vtr, ctxb);

  gemm_bf16_kernel<<<gD, 256, 0, stream>>>(ctxb, wo, xf, nullptr, nullptr, hidden,
                                           NTOK, DMODEL, DMODEL, 0);

  // ---- MLP ----
  rmsnorm_kernel<<<NTOK, 256, 0, stream>>>(xf, ln2, h2b);
  gemm_bf16_kernel<<<gF, 256, 0, stream>>>(h2b, wg, gatef, nullptr, nullptr, nullptr,
                                           NTOK, FFN, DMODEL, 1);   // SiLU
  gemm_bf16_kernel<<<gF, 256, 0, stream>>>(h2b, wu, upf, nullptr, nullptr, nullptr,
                                           NTOK, FFN, DMODEL, 0);
  swiglu_mul_kernel<<<(unsigned)((MF + 255) / 256), 256, 0, stream>>>(gatef, upf, mulb, MF);
  gemm_bf16_kernel<<<gD, 256, 0, stream>>>(mulb, wd, xpost, nullptr, nullptr, xf,
                                           NTOK, DMODEL, FFN, 0);

  // ---- TPN ----
  shift_kernel<<<NTOK, 256, 0, stream>>>(xpost, prevb);
  gemm_bf16_kernel<<<gD, 256, 0, stream>>>(prevb, wt1, t1f, t1b16, t1_b, nullptr,
                                           NTOK, DMODEL, DMODEL, 1); // SiLU
  gemm_bf16_kernel<<<gD, 256, 0, stream>>>(t1b16, wt2, predf, nullptr, t2_b, nullptr,
                                           NTOK, DMODEL, DMODEL, 0);

  // ---- router + losses ----
  router_gemv_kernel<<<NTOK, 256, 0, stream>>>(hidden, r_w, r_b, scoresf);
  tpn_err_kernel<<<NTOK, 256, 0, stream>>>(xpost, hidden, predf, errf, &accums[0]);
  err_stats_kernel<<<1, 256, 0, stream>>>(errf, statsb);
  router_final_kernel<<<NTOK / 256, 256, 0, stream>>>(errf, scoresf, statsb,
                                                      g_o, tgt_o, prob_o, &accums[1]);
  finalize_kernel<<<1, 1, 0, stream>>>(&accums[0], &accums[1], tpn_o, clo_o);
}
